// InteractionNetworkNeighborhood_23158463660311
// MI455X (gfx1250) — matrix operand order repaired and checked by reading, not compiled, not run on hardware
//
#include <hip/hip_runtime.h>
#include <hip/hip_bf16.h>

// ---------------------------------------------------------------------------
// InteractionNetworkNeighborhood edge-MLP for MI455X (gfx1250).
//
// ~86 GFLOP vs ~18 MB unique traffic -> pure matrix-math bound.
//  * Pre-pass converts f32 weights -> f16 once into d_ws, pre-swizzled to the
//    WMMA B-fragment layout (lane%16 = N col; 16 halfs = 16 of the 32 K rows of a chunk).
//  * Main kernel: one wave = TWO queries = two 16-edge x 256-feat A tiles.
//    Each LDS B fragment is loaded once and used by two v_wmma chains
//    (halves DS bytes/FLOP -- LDS read BW [~256B/clk/WGP] is the binding
//    resource when 4 SIMDs stream 1KB B-fragments per WMMA). The two
//    independent accumulator chains also hide WMMA RAW latency.
//  * Weights staged global->LDS via GLOBAL_LOAD_ASYNC_TO_LDS_B128 (ASYNCcnt
//    DMA) per layer phase. LDS: 128KB weights + 8 waves x 16KB act = 256KB.
// ---------------------------------------------------------------------------

typedef __attribute__((ext_vector_type(16))) _Float16 v16h;
typedef __attribute__((ext_vector_type(8)))  _Float16 v8h;
typedef __attribute__((ext_vector_type(8)))  float    v8f;
typedef __attribute__((ext_vector_type(4)))  float    v4f;
typedef __attribute__((ext_vector_type(4)))  int      v4i;

#define BB    4
#define NN    4096
#define KK    16
#define C_IN  128
#define IN_SZ 256
#define H1    256
#define H2    256
#define C_OUT 128

#define WAVES_PER_WG 8
#define QPW          2                       // queries per wave
#define QPWG         (WAVES_PER_WG * QPW)    // 16 queries per workgroup
#define W1_HALFS (IN_SZ * H1)                // 65536
#define W2_HALFS (H1 * H2)                   // 65536
#define W3_HALFS (H2 * C_OUT)                // 32768
#define ACT_HALFS (16 * 256)                 // one query's staging tile
// LDS: weights (128KB) | act (8 waves x 2 x 8KB = 128KB) = 256KB
#define SMEM_BYTES ((W1_HALFS + WAVES_PER_WG * QPW * ACT_HALFS) * 2)

#if defined(__has_builtin)
#if __has_builtin(__builtin_amdgcn_global_load_async_to_lds_b128)
#define ASYNC_LDS_OK 1
#endif
#endif
#ifndef ASYNC_LDS_OK
#define ASYNC_LDS_OK 0
#endif

union AFrag {
    v16h v;
    v8h  h[2];
    _Float16 e[16];
};

// ---------------------------------------------------------------------------
// Pre-pass: convert f32 weights [Kdim, Ndim] row-major -> f16, swizzled into
// WMMA B-fragment order in global scratch. Fragment (nt, kc) = 512 halfs:
//   lane L supplies N column nt*16+L%16; half e -> K = kc*32+e+(e>=8)*8+(L>=16)*8.
// ---------------------------------------------------------------------------
__global__ void __launch_bounds__(256)
swizzle_weights_kernel(const float* __restrict__ W, _Float16* __restrict__ dst,
                       int Kdim, int Ndim) {
    const int KC    = Kdim >> 5;
    const int nblk  = Ndim >> 3;          // power of two (32 or 16)
    const int tasks = Kdim * nblk;
    for (int t = blockIdx.x * 256 + threadIdx.x; t < tasks; t += gridDim.x * 256) {
        const int nb = t & (nblk - 1);
        const int k  = t / nblk;
        const int n0 = nb << 3;
        const float* src = W + (size_t)k * Ndim + n0;
        v4f x0 = *(const v4f*)(src);
        v4f x1 = *(const v4f*)(src + 4);
        const int kc   = k >> 5;
        const int kk   = k & 31;
        const int nt   = n0 >> 4;
        const int e0   = n0 & 15;         // 0 or 8
        const int hb   = (kk >> 3) & 1;
        const int s    = (kk & 7) + ((kk >> 4) << 3);
        _Float16* fb = dst + ((nt * KC + kc) << 9) + s;
#pragma unroll
        for (int j = 0; j < 4; ++j) {
            fb[(e0 + j + hb * 16) << 4]     = (_Float16)x0[j];
            fb[(e0 + 4 + j + hb * 16) << 4] = (_Float16)x1[j];
        }
    }
}

// ---------------------------------------------------------------------------
// Main-kernel helpers
// ---------------------------------------------------------------------------

// Stage pre-swizzled f16 weights global -> LDS (async DMA when available).
__device__ __forceinline__ void stage_ws(const _Float16* __restrict__ src,
                                         _Float16* dst, int halfs) {
    for (int t = threadIdx.x; t < (halfs >> 3); t += 256) {
#if ASYNC_LDS_OK
        __builtin_amdgcn_global_load_async_to_lds_b128(
            (__attribute__((address_space(1))) v4i*)(src + (t << 3)),
            (__attribute__((address_space(3))) v4i*)(dst + (t << 3)),
            0, 0);
#else
        *(v8h*)(dst + (t << 3)) = *(const v8h*)(src + (t << 3));
#endif
    }
}

__device__ __forceinline__ void wait_async_0() {
#if ASYNC_LDS_OK
    asm volatile("s_wait_asynccnt 0x0" ::: "memory");
#endif
}

// Build layer-1 A fragments directly from global feats.
// A layout (16-bit, 16x32): lane L holds row M=L%16; half e -> K = e+(e>=8)*8+(L>=16)*8.
__device__ __forceinline__ void build_l1_afrags(const float* __restrict__ selfp,
                                                const float* __restrict__ nbrp,
                                                AFrag* a) {
    const int l  = threadIdx.x & 31;
    const int hi = l >> 4;
#pragma unroll
    for (int kc = 0; kc < 8; ++kc) {
        const float* src = (kc < 4) ? selfp : nbrp;   // concat(self, neighbor)
        const int cb = (kc & 3) * 32 + hi * 8;
        v4f x0 = *(const v4f*)(src + cb);
        v4f x1 = *(const v4f*)(src + cb + 4);
        v4f x2 = *(const v4f*)(src + cb + 16);
        v4f x3 = *(const v4f*)(src + cb + 20);
#pragma unroll
        for (int j = 0; j < 4; ++j) {
            a[kc].e[j]      = (_Float16)x0[j];
            a[kc].e[4 + j]  = (_Float16)x1[j];
            a[kc].e[8 + j]  = (_Float16)x2[j];
            a[kc].e[12 + j] = (_Float16)x3[j];
        }
    }
}

// Re-load A fragments from a row-major [16 x 256] f16 staging tile.
__device__ __forceinline__ void read_afrags(const _Float16* act, AFrag* a) {
    const int l  = threadIdx.x & 31;
    const int M  = l & 15;
    const int hi = l >> 4;
    const _Float16* row = act + M * 256;
#pragma unroll
    for (int kc = 0; kc < 8; ++kc) {
        const int cb = kc * 32 + hi * 8;
        a[kc].h[0] = *(const v8h*)(row + cb);
        a[kc].h[1] = *(const v8h*)(row + cb + 16);
    }
}

// One dense layer for TWO 16-row A tiles sharing each B fragment:
//   D[16,NT*16] = relu(A x W + b) -> act (f16), for a0/act0 and a1/act1.
// C/D layout: lane l holds column N=l%16; VGPR j holds row M = j + (l>=16)*8.
template <int KC, int NT, bool RELU>
__device__ __forceinline__ void mlp_layer2(const AFrag* a0, const AFrag* a1,
                                           const _Float16* wbuf,
                                           const float* __restrict__ bias,
                                           _Float16* act0, _Float16* act1,
                                           int outW) {
    const int l   = threadIdx.x & 31;
    const int col = l & 15;
    const int hi  = l >> 4;
    for (int nt = 0; nt < NT; ++nt) {
        const float bv = bias[nt * 16 + col];
        v8f acc0 = {bv, bv, bv, bv, bv, bv, bv, bv};
        v8f acc1 = acc0;
#pragma unroll
        for (int kc = 0; kc < KC; ++kc) {
            v16h bf = *(const v16h*)(wbuf + (((nt * KC + kc) << 9) + (l << 4)));
            acc0 = __builtin_amdgcn_wmma_f32_16x16x32_f16(
                false, a0[kc].v, false, bf, (short)0, acc0, false, false);
            acc1 = __builtin_amdgcn_wmma_f32_16x16x32_f16(
                false, a1[kc].v, false, bf, (short)0, acc1, false, false);
        }
#pragma unroll
        for (int j = 0; j < 8; ++j) {
            float v0 = acc0[j], v1 = acc1[j];
            if (RELU) { v0 = fmaxf(v0, 0.0f); v1 = fmaxf(v1, 0.0f); }
            const int o = (j + hi * 8) * outW + nt * 16 + col;
            act0[o] = (_Float16)v0;
            act1[o] = (_Float16)v1;
        }
    }
}

// Final layer (256 -> 128, +b3) for two queries, fused with the masked
// reduction over the 16 neighbor rows. Lanes l and l+16 hold the same column
// for the two row halves -> one shfl_xor(16) completes each K-sum.
__device__ __forceinline__ void mlp_layer3_reduce2(const AFrag* a0, const AFrag* a1,
                                                   const _Float16* wbuf,
                                                   const float* __restrict__ b3,
                                                   const int* __restrict__ valid0,
                                                   const int* __restrict__ valid1,
                                                   float* __restrict__ out0,
                                                   float* __restrict__ out1) {
    const int l   = threadIdx.x & 31;
    const int col = l & 15;
    const int hi  = l >> 4;
    float m0[8], m1[8];
#pragma unroll
    for (int j = 0; j < 8; ++j) {
        m0[j] = (float)valid0[j + hi * 8];
        m1[j] = (float)valid1[j + hi * 8];
    }
    for (int nt = 0; nt < 8; ++nt) {
        const float bv = b3[nt * 16 + col];
        v8f acc0 = {bv, bv, bv, bv, bv, bv, bv, bv};
        v8f acc1 = acc0;
#pragma unroll
        for (int kc = 0; kc < 8; ++kc) {
            v16h bf = *(const v16h*)(wbuf + (((nt * 8 + kc) << 9) + (l << 4)));
            acc0 = __builtin_amdgcn_wmma_f32_16x16x32_f16(
                false, a0[kc].v, false, bf, (short)0, acc0, false, false);
            acc1 = __builtin_amdgcn_wmma_f32_16x16x32_f16(
                false, a1[kc].v, false, bf, (short)0, acc1, false, false);
        }
        float s0 = 0.0f, s1 = 0.0f;
#pragma unroll
        for (int j = 0; j < 8; ++j) {
            s0 += acc0[j] * m0[j];
            s1 += acc1[j] * m1[j];
        }
        s0 += __shfl_xor(s0, 16, 32);
        s1 += __shfl_xor(s1, 16, 32);
        if (l < 16) {
            out0[nt * 16 + col] = s0;
            out1[nt * 16 + col] = s1;
        }
    }
}

// ---------------------------------------------------------------------------
// Main kernel
// ---------------------------------------------------------------------------
__global__ void __launch_bounds__(256, 1)
edge_mlp_wmma_kernel(const float* __restrict__ feats,
                     const int* __restrict__ n_idxs,
                     const int* __restrict__ neighbor_valid,
                     const _Float16* __restrict__ ws,   // pre-swizzled f16 weights
                     const float* __restrict__ b1,
                     const float* __restrict__ b2,
                     const float* __restrict__ b3,
                     float* __restrict__ out) {
    extern __shared__ __align__(16) unsigned char smem[];
    _Float16* wbuf = (_Float16*)smem;                  // 65536 halfs (128KB)
    const int wave = threadIdx.x >> 5;
    _Float16* act0 = wbuf + W1_HALFS + (wave * QPW) * ACT_HALFS;
    _Float16* act1 = act0 + ACT_HALFS;

    const _Float16* wsW1 = ws;
    const _Float16* wsW2 = ws + W1_HALFS;
    const _Float16* wsW3 = ws + W1_HALFS + W2_HALFS;

    const int qg0 = blockIdx.x * QPWG + wave * QPW;    // two query ids per wave
    const int qg1 = qg0 + 1;
    const int b0  = qg0 >> 12;                         // / NN
    const int b1i = qg1 >> 12;
    const int l   = threadIdx.x & 31;
    const int M   = l & 15;                            // neighbor slot (row)

    // ---- phase 1: DMA W1, layer 1 from global gathers ----
    stage_ws(wsW1, wbuf, W1_HALFS);
    wait_async_0();
    __syncthreads();

    const float* self0 = feats + (size_t)qg0 * C_IN;
    const float* self1 = feats + (size_t)qg1 * C_IN;
    const int idx0     = n_idxs[qg0 * KK + M];
    const int idx1     = n_idxs[qg1 * KK + M];
    const float* nbr0  = feats + ((size_t)(b0  * NN + idx0)) * C_IN;
    const float* nbr1  = feats + ((size_t)(b1i * NN + idx1)) * C_IN;

    AFrag a0[8], a1[8];
    build_l1_afrags(self0, nbr0, a0);
    build_l1_afrags(self1, nbr1, a1);
    mlp_layer2<8, 16, true>(a0, a1, wbuf, b1, act0, act1, 256);

    // ---- phase 2: DMA W2, layer 2 ----
    __syncthreads();                   // all readers of W1 done
    stage_ws(wsW2, wbuf, W2_HALFS);
    wait_async_0();
    __syncthreads();
    read_afrags(act0, a0);
    read_afrags(act1, a1);
    mlp_layer2<8, 16, true>(a0, a1, wbuf, b2, act0, act1, 256);

    // ---- phase 3: DMA W3, layer 3 + masked neighbor reduction ----
    __syncthreads();
    stage_ws(wsW3, wbuf, W3_HALFS);
    wait_async_0();
    __syncthreads();
    read_afrags(act0, a0);
    read_afrags(act1, a1);
    mlp_layer3_reduce2(a0, a1, wbuf, b3,
                       neighbor_valid + (size_t)qg0 * KK,
                       neighbor_valid + (size_t)qg1 * KK,
                       out + (size_t)qg0 * C_OUT,
                       out + (size_t)qg1 * C_OUT);
}

extern "C" void kernel_launch(void* const* d_in, const int* in_sizes, int n_in,
                              void* d_out, int out_size, void* d_ws, size_t ws_size,
                              hipStream_t stream) {
    // setup_inputs order: keys(0), points(1), feats(2), n_idxs(3),
    // neighbor_valid(4), W1(5), b1(6), W2(7), b2(8), W3(9), b3(10).
    // keys/points are unused by the reference computation.
    const float* feats = (const float*)d_in[2];
    const int*   nidx  = (const int*)d_in[3];
    const int*   valid = (const int*)d_in[4];
    const float* W1 = (const float*)d_in[5];
    const float* b1 = (const float*)d_in[6];
    const float* W2 = (const float*)d_in[7];
    const float* b2 = (const float*)d_in[8];
    const float* W3 = (const float*)d_in[9];
    const float* b3 = (const float*)d_in[10];
    float* out = (float*)d_out;

    // d_ws: 320KB of pre-swizzled f16 weights (W1 | W2 | W3).
    _Float16* ws = (_Float16*)d_ws;

    // Pre-pass: convert + swizzle weights once (stream-ordered before main).
    swizzle_weights_kernel<<<32, 256, 0, stream>>>(W1, ws, IN_SZ, H1);
    swizzle_weights_kernel<<<32, 256, 0, stream>>>(W2, ws + W1_HALFS, H1, H2);
    swizzle_weights_kernel<<<16, 256, 0, stream>>>(W3, ws + W1_HALFS + W2_HALFS, H2, C_OUT);

    const int n_queries = BB * NN;                     // 16384
    dim3 grid(n_queries / QPWG);                       // 1024
    dim3 block(256);
    edge_mlp_wmma_kernel<<<grid, block, SMEM_BYTES, stream>>>(
        feats, nidx, valid, ws, b1, b2, b3, out);
}
